// GeoRound_8186207666917
// MI455X (gfx1250) — matrix operand order repaired and checked by reading, not compiled, not run on hardware
//
#include <hip/hip_runtime.h>
#include <stdint.h>

#define D_BLADE 256
#define D_FFN   1024
#define N_TOK   8192
#define NROWS   (N_TOK * 8)

typedef __attribute__((ext_vector_type(16))) __bf16 v16bf;
typedef __attribute__((ext_vector_type(8)))  float  v8f;

union Frag { uint4 q[2]; v16bf v; };

// Cayley table target indices k for p = i*8+j, derived from the reference's
// _derive_cayley_table() (Cl(3,0): 1,e1,e2,e3,e12,e13,e23,e123). Signs come
// from the sg input at runtime.
constexpr int TK[64] = {
  0,1,2,3,4,5,6,7,
  1,0,4,5,2,3,7,6,
  2,4,0,6,1,7,3,5,
  3,5,6,0,7,1,2,4,
  4,2,1,7,0,6,5,3,
  5,3,7,1,6,0,4,2,
  6,7,3,2,5,4,0,1,
  7,6,5,4,3,2,1,0
};

__device__ __forceinline__ unsigned short f32_bf16(float f) {
  unsigned int u = __float_as_uint(f);
  u += 0x7fffu + ((u >> 16) & 1u);   // round-to-nearest-even
  return (unsigned short)(u >> 16);
}

__device__ __forceinline__ float sigm(float v) {
  return 1.0f / (1.0f + __expf(-v));
}

__global__ void cvt_w_bf16(const float* __restrict__ gw, const float* __restrict__ uw,
                           const float* __restrict__ dw, uint16_t* __restrict__ o) {
  const int NW = D_FFN * D_BLADE;
  for (int i = blockIdx.x * blockDim.x + threadIdx.x; i < 3 * NW;
       i += gridDim.x * blockDim.x) {
    float v = (i < NW) ? gw[i] : (i < 2 * NW ? uw[i - NW] : dw[i - 2 * NW]);
    o[i] = f32_bf16(v);
  }
}

__global__ __launch_bounds__(256)
void geo_ffn_kernel(const float* __restrict__ x,   const float* __restrict__ iw,
                    const float* __restrict__ sg,  const float* __restrict__ gg,
                    const float* __restrict__ lnw, const float* __restrict__ lnb,
                    const uint16_t* __restrict__ gate_bf,
                    const uint16_t* __restrict__ up_bf,
                    const uint16_t* __restrict__ down_bf,
                    float* __restrict__ out) {
  // LDS layout (56.3 KB): xs 16KB | Hb 32KB (aliased by mixb 16KB) | normA 8KB | sgw 256B
  __shared__ __align__(16) unsigned char smem[16384 + 32768 + 8192 + 256];
  float*    xs    = (float*)smem;
  uint16_t* Hb    = (uint16_t*)(smem + 16384);
  float*    mixb  = (float*)(smem + 16384);            // alias: dead before Hb is written
  uint16_t* normA = (uint16_t*)(smem + 16384 + 32768);
  float*    sgw   = (float*)(smem + 16384 + 32768 + 8192);

  const int tid  = threadIdx.x;
  const int lane = tid & 31;
  const int wid  = tid >> 5;
  const size_t rowbase = (size_t)blockIdx.x * 16;      // 16 rows = 2 tokens

  // ---- Phase 0: load x tile (16 x 256 f32) and per-pair weights ----
  {
    const float4* xg  = (const float4*)(x + rowbase * D_BLADE);
    float4*       xs4 = (float4*)xs;
#pragma unroll
    for (int q = 0; q < 4; ++q) xs4[tid + q * 256] = xg[tid + q * 256];
  }
  if (tid < 64) sgw[tid] = sg[tid] * sigm(iw[tid]);
  __syncthreads();

  const float g = sigm(gg[0]);

  // ---- Phase 1: geometric product + blend (thread tid owns column d=tid, both tokens) ----
#pragma unroll
  for (int t = 0; t < 2; ++t) {
    float xv[8], geo[8];
#pragma unroll
    for (int b = 0; b < 8; ++b) { xv[b] = xs[(t * 8 + b) * D_BLADE + tid]; geo[b] = 0.0f; }
#pragma unroll
    for (int p = 0; p < 64; ++p)
      geo[TK[p]] = fmaf(sgw[p] * xv[p >> 3], xv[p & 7], geo[TK[p]]);
#pragma unroll
    for (int b = 0; b < 8; ++b)
      mixb[(t * 8 + b) * D_BLADE + tid] = g * geo[b] + (1.0f - g) * xv[b];
  }
  __syncthreads();

  // ---- Phase 2: LayerNorm per row (each wave: 2 rows), emit bf16 A-tile ----
#pragma unroll
  for (int rr = 0; rr < 2; ++rr) {
    const int row = wid * 2 + rr;
    float v[8], s = 0.0f, s2 = 0.0f;
#pragma unroll
    for (int q = 0; q < 8; ++q) {
      v[q] = mixb[row * D_BLADE + lane + q * 32];
      s += v[q]; s2 += v[q] * v[q];
    }
#pragma unroll
    for (int off = 16; off >= 1; off >>= 1) {
      s  += __shfl_xor(s,  off, 32);
      s2 += __shfl_xor(s2, off, 32);
    }
    const float mu = s * (1.0f / 256.0f);
    const float rs = rsqrtf(s2 * (1.0f / 256.0f) - mu * mu + 1e-5f);
#pragma unroll
    for (int q = 0; q < 8; ++q) {
      const int d = lane + q * 32;
      normA[row * D_BLADE + d] = f32_bf16((v[q] - mu) * rs * lnw[d] + lnb[d]);
    }
  }
  __syncthreads();

  // Fragment addressing (wave32; this code's own contiguous K order, same in A and B):
  //  A (16x32 bf16): lane L -> row L&15; contiguous 16 K at k (lo lanes) / k+16 (hi lanes)
  //  B (32x16 bf16): lane L -> col L&15; contiguous 16 K at k (lo lanes) / k+16 (hi lanes)
  const int arow = lane & 15;
  const int hi   = lane >> 4;
  const int aoff = hi * 16;
  const int boff = hi * 16;

  // ---- Phase 3: GEMM1 gate/up (K=256) + SiLU*up -> bf16 H tile in LDS ----
  for (int ff = 0; ff < 8; ++ff) {
    const int ft = wid * 8 + ff;            // wave owns 8 F-tiles -> 128 cols of H
    const int fn = ft * 16 + arow;
    const uint16_t* gb = gate_bf + fn * D_BLADE;
    const uint16_t* ub = up_bf   + fn * D_BLADE;
    const uint16_t* ab = normA + arow * D_BLADE;
    v8f cg = {}; v8f cu = {};
#pragma unroll
    for (int kt = 0; kt < 8; ++kt) {
      Frag a, b1, b2;
      const uint4* ap = (const uint4*)(ab + kt * 32 + aoff);
      a.q[0] = ap[0]; a.q[1] = ap[1];
      const uint4* gp = (const uint4*)(gb + kt * 32 + boff);
      b1.q[0] = gp[0]; b1.q[1] = gp[1];
      const uint4* up = (const uint4*)(ub + kt * 32 + boff);
      b2.q[0] = up[0]; b2.q[1] = up[1];
      cg = __builtin_amdgcn_wmma_f32_16x16x32_bf16(false, a.v, false, b1.v, (short)0, cg, false, false);
      cu = __builtin_amdgcn_wmma_f32_16x16x32_bf16(false, a.v, false, b2.v, (short)0, cu, false, false);
    }
#pragma unroll
    for (int r = 0; r < 8; ++r) {
      const float gv = cg[r];
      const float h  = (gv * sigm(gv)) * cu[r];
      Hb[(hi * 8 + r) * D_FFN + ft * 16 + arow] = f32_bf16(h);
    }
  }
  __syncthreads();

  // ---- Phase 4: GEMM2 down (K=1024) + residual ----
#pragma unroll
  for (int nn = 0; nn < 2; ++nn) {
    const int nt   = wid * 2 + nn;          // wave owns 2 of 16 N-tiles
    const int dcol = nt * 16 + arow;
    const uint16_t* hb = Hb + arow * D_FFN;
    const uint16_t* db = down_bf + dcol * D_FFN;
    v8f c = {};
#pragma unroll 4
    for (int kt = 0; kt < 32; ++kt) {
      Frag a, b;
      const uint4* ap = (const uint4*)(hb + kt * 32 + aoff);
      a.q[0] = ap[0]; a.q[1] = ap[1];
      const uint4* bp = (const uint4*)(db + kt * 32 + boff);
      b.q[0] = bp[0]; b.q[1] = bp[1];
      c = __builtin_amdgcn_wmma_f32_16x16x32_bf16(false, a.v, false, b.v, (short)0, c, false, false);
    }
#pragma unroll
    for (int r = 0; r < 8; ++r) {
      const int m = hi * 8 + r;
      out[(rowbase + m) * D_BLADE + dcol] = xs[m * D_BLADE + dcol] + c[r];
    }
  }
}

extern "C" void kernel_launch(void* const* d_in, const int* in_sizes, int n_in,
                              void* d_out, int out_size, void* d_ws, size_t ws_size,
                              hipStream_t stream) {
  (void)in_sizes; (void)n_in; (void)out_size; (void)ws_size;
  const float* x   = (const float*)d_in[0];
  const float* iw  = (const float*)d_in[1];
  const float* sg  = (const float*)d_in[2];
  const float* gg  = (const float*)d_in[3];
  const float* gw  = (const float*)d_in[4];
  const float* uw  = (const float*)d_in[5];
  const float* dw  = (const float*)d_in[6];
  const float* lnw = (const float*)d_in[7];
  const float* lnb = (const float*)d_in[8];
  // d_in[9..11] (si, sj, tk) are deterministic; baked into the kernel.

  uint16_t* gate_bf = (uint16_t*)d_ws;                 // 512 KB
  uint16_t* up_bf   = gate_bf + D_FFN * D_BLADE;       // 512 KB
  uint16_t* down_bf = up_bf   + D_FFN * D_BLADE;       // 512 KB (contiguous conversion)

  cvt_w_bf16<<<192, 256, 0, stream>>>(gw, uw, dw, gate_bf);
  geo_ffn_kernel<<<NROWS / 16, 256, 0, stream>>>(x, iw, sg, gg, lnw, lnb,
                                                 gate_bf, up_bf, down_bf,
                                                 (float*)d_out);
}
